// TiledNN_43009802502531
// MI455X (gfx1250) — compile-verified, matrix operand order checked by reading, not run on hardware
//
#include <hip/hip_runtime.h>
#include <hip/hip_bf16.h>

typedef __attribute__((ext_vector_type(16))) _Float16 v16h;
typedef __attribute__((ext_vector_type(8)))  _Float16 v8h;
typedef __attribute__((ext_vector_type(8)))  float    v8f;

#define HIDDEN  4096
#define IN_DIM  784
#define KPAD    800      // 784 padded to 25*32
#define OUT_DIM 10
#define NPAD2   16       // 10 padded to 16
#define BATCH   8192
#define CF      4

// ---------------------------------------------------------------------------
// Prep kernels: dequantize weights to f16 (zero padded), convert x to f16.
// ---------------------------------------------------------------------------
__global__ __launch_bounds__(256) void prep_x_f16(const float* __restrict__ x,
                                                  _Float16* __restrict__ xf) {
  int idx = blockIdx.x * 256 + threadIdx.x;           // over BATCH*KPAD
  int row = idx / KPAD, col = idx % KPAD;
  float v = (col < IN_DIM) ? x[row * IN_DIM + col] : 0.0f;
  xf[idx] = (_Float16)v;
}

__global__ __launch_bounds__(256) void prep_w1_f16(const int* __restrict__ tile1,
                                                   const float* __restrict__ alphas1,
                                                   _Float16* __restrict__ w1f) {
  int idx = blockIdx.x * 256 + threadIdx.x;           // over HIDDEN*KPAD
  int n = idx / KPAD, k = idx % KPAD;
  float v = 0.0f;
  if (k < IN_DIM) {
    // w1[n,k] = alphas1[n/1024] * (2*tile1[(n%1024)*784 + k] - 1)
    int chunk = n >> 10;
    int s = 2 * tile1[(n & 1023) * IN_DIM + k] - 1;
    v = alphas1[chunk] * (float)s;
  }
  w1f[idx] = (_Float16)v;
}

__global__ __launch_bounds__(256) void prep_w2_f16(const int* __restrict__ tile2,
                                                   const float* __restrict__ alphas2,
                                                   _Float16* __restrict__ w2f) {
  int idx = blockIdx.x * 256 + threadIdx.x;           // over NPAD2*HIDDEN
  int n = idx / HIDDEN, k = idx % HIDDEN;
  float v = 0.0f;
  if (n < OUT_DIM) {
    int f = n * HIDDEN + k;                           // flat index into w2
    int chunk = f / (OUT_DIM * HIDDEN / CF);          // /10240
    int s = 2 * tile2[f % (OUT_DIM * HIDDEN / CF)] - 1;
    v = alphas2[chunk] * (float)s;
  }
  w2f[idx] = (_Float16)v;
}

// ---------------------------------------------------------------------------
// Fragment pack helper: elements 0-7 = K[k0..k0+7], 8-15 = K[k0+16..k0+23]
// (matches 16-bit A/B VGPR layout for v_wmma_*_16x16x32 on wave32).
// ---------------------------------------------------------------------------
__device__ inline v16h pack16(v8h lo, v8h hi) {
  v16h o;
#pragma unroll
  for (int i = 0; i < 8; ++i) { o[i] = lo[i]; o[i + 8] = hi[i]; }
  return o;
}

// ---------------------------------------------------------------------------
// GEMM1 + ReLU:  H[8192,4096] = relu(X[8192,800] @ W1[4096,800]^T), f16 out.
// Block: 256 thr (8 waves, 2(M)x4(N)), tile 128x128, K-step 32, LDS staged.
// ---------------------------------------------------------------------------
__global__ __launch_bounds__(256) void gemm1_relu(const _Float16* __restrict__ X,
                                                  const _Float16* __restrict__ W1,
                                                  _Float16* __restrict__ H) {
  __shared__ _Float16 As[128 * 32];
  __shared__ _Float16 Bs[128 * 32];

  const int tid   = threadIdx.x;
  const int lane  = tid & 31;
  const int wave  = tid >> 5;
  const int waveM = wave >> 2;                 // 0..1  -> 64 rows each
  const int waveN = wave & 3;                  // 0..3  -> 32 cols each
  const int mBlock = blockIdx.y * 128;
  const int nBlock = blockIdx.x * 128;

  const int kh = lane >> 4;                    // lane half (K phase)
  const int fr = lane & 15;                    // fragment row

  v8f acc[4][2] = {};

  for (int ks = 0; ks < KPAD / 32; ++ks) {
    const int kBase = ks * 32;

    // Cooperative stage: 128x32 f16 tile = 512 x v8h chunks; 2 chunks/thread.
#pragma unroll
    for (int j = 0; j < 2; ++j) {
      int c   = tid + 256 * j;
      int row = c >> 2;
      int col = (c & 3) * 8;
      *(v8h*)&As[row * 32 + col] =
          *(const v8h*)&X[(size_t)(mBlock + row) * KPAD + kBase + col];
      *(v8h*)&Bs[row * 32 + col] =
          *(const v8h*)&W1[(size_t)(nBlock + row) * KPAD + kBase + col];
    }
    // Prefetch next K tile of X into cache while we compute this one.
    if (ks + 1 < KPAD / 32)
      __builtin_prefetch(&X[(size_t)(mBlock + (tid >> 2)) * KPAD + kBase + 32], 0, 0);
    __syncthreads();

    // Per-wave fragments out of LDS.
    v16h a[4], b[2];
#pragma unroll
    for (int mt = 0; mt < 4; ++mt) {
      const _Float16* p = &As[(waveM * 64 + mt * 16 + fr) * 32 + kh * 8];
      a[mt] = pack16(*(const v8h*)p, *(const v8h*)(p + 16));
    }
#pragma unroll
    for (int nt = 0; nt < 2; ++nt) {
      const _Float16* p = &Bs[(waveN * 32 + nt * 16 + fr) * 32 + kh * 8];
      b[nt] = pack16(*(const v8h*)p, *(const v8h*)(p + 16));
    }

#pragma unroll
    for (int mt = 0; mt < 4; ++mt)
#pragma unroll
      for (int nt = 0; nt < 2; ++nt)
        acc[mt][nt] = __builtin_amdgcn_wmma_f32_16x16x32_f16(
            false, a[mt], false, b[nt], (short)0, acc[mt][nt], false, false);

    __syncthreads();
  }

  // Epilogue: ReLU, convert to f16, store.
  // C layout: VGPR r -> lanes 0-15: M=r, lanes 16-31: M=8+r; N = lane&15.
  const int nlane = lane & 15;
#pragma unroll
  for (int mt = 0; mt < 4; ++mt) {
#pragma unroll
    for (int nt = 0; nt < 2; ++nt) {
      int col = nBlock + waveN * 32 + nt * 16 + nlane;
#pragma unroll
      for (int r = 0; r < 8; ++r) {
        int row = mBlock + waveM * 64 + mt * 16 + kh * 8 + r;
        float v = acc[mt][nt][r];
        v = v > 0.0f ? v : 0.0f;
        H[(size_t)row * HIDDEN + col] = (_Float16)v;
      }
    }
  }
}

// ---------------------------------------------------------------------------
// GEMM2:  OUT[8192,10] = H[8192,4096] @ W2[16,4096]^T (cols 10..15 padded 0).
// One 16-row M tile per wave; W2 (128 KB) is L2 resident.
// ---------------------------------------------------------------------------
__global__ __launch_bounds__(256) void gemm2(const _Float16* __restrict__ H,
                                             const _Float16* __restrict__ W2,
                                             float* __restrict__ OUT) {
  const int tid  = threadIdx.x;
  const int lane = tid & 31;
  const int wave = tid >> 5;
  const int mBase = (blockIdx.x * 8 + wave) * 16;

  const int kh = lane >> 4;
  const int fr = lane & 15;

  v8f acc = {};

  for (int ks = 0; ks < HIDDEN / 32; ++ks) {
    const int kBase = ks * 32;
    const _Float16* pa = H + (size_t)(mBase + fr) * HIDDEN + kBase + kh * 8;
    const _Float16* pb = W2 + (size_t)fr * HIDDEN + kBase + kh * 8;
    v16h a = pack16(*(const v8h*)pa, *(const v8h*)(pa + 16));
    v16h b = pack16(*(const v8h*)pb, *(const v8h*)(pb + 16));
    acc = __builtin_amdgcn_wmma_f32_16x16x32_f16(false, a, false, b,
                                                 (short)0, acc, false, false);
  }

  const int col = lane & 15;
  if (col < OUT_DIM) {
#pragma unroll
    for (int r = 0; r < 8; ++r) {
      int row = mBase + kh * 8 + r;
      OUT[(size_t)row * OUT_DIM + col] = acc[r];
    }
  }
}

// ---------------------------------------------------------------------------
// Launch
// ---------------------------------------------------------------------------
extern "C" void kernel_launch(void* const* d_in, const int* in_sizes, int n_in,
                              void* d_out, int out_size, void* d_ws, size_t ws_size,
                              hipStream_t stream) {
  const float* x       = (const float*)d_in[0];
  const int*   tile1   = (const int*)d_in[1];
  const float* alphas1 = (const float*)d_in[2];
  const int*   tile2   = (const int*)d_in[3];
  const float* alphas2 = (const float*)d_in[4];
  float* out = (float*)d_out;

  // Workspace layout (f16 buffers, all 256B-aligned offsets):
  //  xf16 [8192][800]  : 13,107,200 B
  //  w1f  [4096][800]  :  6,553,600 B
  //  w2f  [  16][4096] :    131,072 B
  //  h    [8192][4096] : 67,108,864 B   (total ~86.9 MB)
  char* base = (char*)d_ws;
  _Float16* xf  = (_Float16*)(base);
  _Float16* w1f = (_Float16*)(base + 13107200);
  _Float16* w2f = (_Float16*)(base + 13107200 + 6553600);
  _Float16* h   = (_Float16*)(base + 13107200 + 6553600 + 131072);

  // Prep
  prep_x_f16 <<<(BATCH  * KPAD)   / 256, 256, 0, stream>>>(x, xf);
  prep_w1_f16<<<(HIDDEN * KPAD)   / 256, 256, 0, stream>>>(tile1, alphas1, w1f);
  prep_w2_f16<<<(NPAD2  * HIDDEN) / 256, 256, 0, stream>>>(tile2, alphas2, w2f);

  // Layer 1: 128x128 tiles -> grid (4096/128, 8192/128)
  dim3 g1(HIDDEN / 128, BATCH / 128);
  gemm1_relu<<<g1, 256, 0, stream>>>(xf, w1f, h);

  // Layer 2: 8 waves * 16 rows = 128 rows/block
  gemm2<<<BATCH / 128, 256, 0, stream>>>(h, w2f, out);
}
